// ScaledDotProductAttention_15264313770258
// MI455X (gfx1250) — matrix operand order repaired and checked by reading, not compiled, not run on hardware
//
#include <hip/hip_runtime.h>

// Causal attention (B=2,H=16,S=2048,D=64), fp32 I/O, outputs (out, attn).
// Two-pass flash-style per 16-row Q tile per wave32, bf16 WMMA matmuls.
// v3: attn written as b128 non-temporal stores via the LDS-staged P tile.

#define B_N 2
#define H_N 16
#define S_LEN 2048
#define D_DIM 64
#define NEGV (-1e9f)

typedef __attribute__((ext_vector_type(16))) __bf16 v16bf;
typedef __attribute__((ext_vector_type(2)))  __bf16 bf16x2;
typedef __attribute__((ext_vector_type(8)))  float  v8f;
typedef __attribute__((ext_vector_type(4)))  float  f32x4;

// Pack two f32 -> bf16x2. Single-instruction v_cvt_pk_bf16_f32 (probe-confirmed on gfx1250).
static __device__ __forceinline__ bf16x2 pk_bf16(float x, float y) {
#if __has_builtin(__builtin_amdgcn_cvt_pk_bf16_f32)
  return __builtin_amdgcn_cvt_pk_bf16_f32(x, y);
#else
  bf16x2 r;
  r[0] = (__bf16)x;
  r[1] = (__bf16)y;
  return r;
#endif
}

// reductions across the 16 lanes of a half-wave (masks 1,2,4,8 stay in-half)
static __device__ __forceinline__ float half_max(float v) {
#pragma unroll
  for (int m = 1; m <= 8; m <<= 1) v = fmaxf(v, __shfl_xor(v, m, 32));
  return v;
}
static __device__ __forceinline__ float half_sum(float v) {
#pragma unroll
  for (int m = 1; m <= 8; m <<= 1) v += __shfl_xor(v, m, 32);
  return v;
}

// 16-bit B fragment (32 K x 16 N): element 2*vv+{0,1} <-> K = 16*(lane/16)+2*vv+{0,1}
static __device__ __forceinline__ v16bf load_b_pair(const float* p) {
  v16bf b;
#pragma unroll
  for (int vv = 0; vv < 8; ++vv) {
    float2 t = *(const float2*)(p + 2 * vv);
    bf16x2 c = pk_bf16(t.x, t.y);
    b[2 * vv]     = c[0];
    b[2 * vv + 1] = c[1];
  }
  return b;
}

// 16-bit A fragment (16 M x 32 K): element 2*vv+e <-> K = 16*(lane/16)+2*vv+e
static __device__ __forceinline__ v16bf load_a_frag(const float* p, float scale) {
  v16bf a;
#pragma unroll
  for (int vv = 0; vv < 8; ++vv) {
    float2 t = *(const float2*)(p + 2 * vv);
    bf16x2 c = pk_bf16(t.x * scale, t.y * scale);
    a[2 * vv]     = c[0];
    a[2 * vv + 1] = c[1];
  }
  return a;
}

// V as B fragment: element i <-> key offset 16*(lane/16)+i, column d = lane%16 (+dblk*16)
static __device__ __forceinline__ v16bf load_v_frag(const float* p) {
  v16bf b;
#pragma unroll
  for (int vv = 0; vv < 8; ++vv) {
    bf16x2 c = pk_bf16(p[(2 * vv) * D_DIM], p[(2 * vv + 1) * D_DIM]);
    b[2 * vv]     = c[0];
    b[2 * vv + 1] = c[1];
  }
  return b;
}

__global__ __launch_bounds__(256) void fa_causal_kernel(
    const float* __restrict__ q, const float* __restrict__ k,
    const float* __restrict__ v, float* __restrict__ outp,
    float* __restrict__ attn) {
  __shared__ float plds[8][16 * 32];  // per-wave 16x32 P staging (2 KB/wave)

  const int lane = threadIdx.x & 31;
  const int wv   = threadIdx.x >> 5;
  const int tile = blockIdx.x * 8 + wv;        // 4096 q-tiles total
  const int bh   = tile >> 7;                  // 128 tiles per (b,h)
  const int qb   = (tile & 127) << 4;          // q row base
  const int n    = lane & 15;
  const int h4   = lane >> 4;
  const size_t bhS = (size_t)bh * S_LEN;

  // b128 store geometry for the 16x32 attn tile: 4 float4s per lane
  //   idx = lane + 32*i  ->  row = idx/8 (0..15), col group = (idx%8)*4
  // Resident Q A-fragments, pre-scaled by 1/sqrt(D)=0.125.
  const float* qp = q + (bhS + qb + n) * D_DIM + 16 * h4;
  const v16bf qa0 = load_a_frag(qp, 0.125f);
  const v16bf qa1 = load_a_frag(qp + 32, 0.125f);

  const float* kbase_bh = k + bhS * D_DIM;
  const float* vbase_bh = v + bhS * D_DIM;
  const int ktmax = (qb + 15) >> 5;  // last 32-key step with any valid key

  // ---- Pass 1: lane-local online softmax stats (no per-step reductions) ----
  float m[8], l[8];
#pragma unroll
  for (int r = 0; r < 8; ++r) { m[r] = -1e30f; l[r] = 0.0f; }

  for (int kt = 0; kt <= ktmax; ++kt) {
    const int kt32 = kt << 5;
    if (kt < ktmax) {  // prefetch next K tile -> gfx1250 global_prefetch
      const float* kpf = kbase_bh + (size_t)(kt32 + 32 + n) * D_DIM + 16 * h4;
      __builtin_prefetch(kpf, 0, 0);
      __builtin_prefetch(kpf + 16 * D_DIM, 0, 0);
    }
    v8f sc[2];
#pragma unroll
    for (int s2 = 0; s2 < 2; ++s2) {
      const float* kb = kbase_bh + (size_t)(kt32 + s2 * 16 + n) * D_DIM + 16 * h4;
      v8f acc = {};
      acc = __builtin_amdgcn_wmma_f32_16x16x32_bf16(false, qa0, false, load_b_pair(kb),      (short)0, acc, false, false);
      acc = __builtin_amdgcn_wmma_f32_16x16x32_bf16(false, qa1, false, load_b_pair(kb + 32), (short)0, acc, false, false);
      sc[s2] = acc;
    }
#pragma unroll
    for (int r = 0; r < 8; ++r) {
      const int qidx = qb + r + 8 * h4;
      const float s0 = (kt32 + n      > qidx) ? NEGV : sc[0][r];
      const float s1 = (kt32 + 16 + n > qidx) ? NEGV : sc[1][r];
      const float mn = fmaxf(m[r], fmaxf(s0, s1));
      l[r] = l[r] * __expf(m[r] - mn) + __expf(s0 - mn) + __expf(s1 - mn);
      m[r] = mn;
    }
  }
  // merge lane-local stats across the 16 lanes holding each row
  float linv[8];
#pragma unroll
  for (int r = 0; r < 8; ++r) {
    const float mr = half_max(m[r]);
    const float ls = half_sum(l[r] * __expf(m[r] - mr));  // masked-only lanes zero out
    m[r] = mr;
    linv[r] = 1.0f / ls;
  }

  // ---- Pass 2: normalized probabilities -> attn (b128), O += P*V ----
  v8f o[4] = {};
  float* lp = &plds[wv][0];

  for (int kt = 0; kt < S_LEN / 32; ++kt) {
    const int kt32 = kt << 5;
    if (kt <= ktmax) {
      if (kt < ktmax) {  // prefetch next K and V tiles
        const float* kpf = kbase_bh + (size_t)(kt32 + 32 + n) * D_DIM + 16 * h4;
        __builtin_prefetch(kpf, 0, 0);
        __builtin_prefetch(kpf + 16 * D_DIM, 0, 0);
        const float* vpf = vbase_bh + (size_t)(kt32 + 32 + 16 * h4) * D_DIM + n;
        __builtin_prefetch(vpf, 0, 0);
      }
      v8f sc[2];
#pragma unroll
      for (int s2 = 0; s2 < 2; ++s2) {
        const float* kb = kbase_bh + (size_t)(kt32 + s2 * 16 + n) * D_DIM + 16 * h4;
        v8f acc = {};
        acc = __builtin_amdgcn_wmma_f32_16x16x32_bf16(false, qa0, false, load_b_pair(kb),      (short)0, acc, false, false);
        acc = __builtin_amdgcn_wmma_f32_16x16x32_bf16(false, qa1, false, load_b_pair(kb + 32), (short)0, acc, false, false);
        sc[s2] = acc;
      }
      // probabilities -> LDS (row-major 16x32 tile)
#pragma unroll
      for (int s2 = 0; s2 < 2; ++s2) {
#pragma unroll
        for (int r = 0; r < 8; ++r) {
          const int qidx = qb + r + 8 * h4;
          const int kidx = kt32 + s2 * 16 + n;
          const float p = (kidx > qidx) ? 0.0f : __expf(sc[s2][r] - m[r]) * linv[r];
          lp[(r + 8 * h4) * 32 + s2 * 16 + n] = p;
        }
      }
      asm volatile("s_wait_dscnt 0x0" ::: "memory");  // LDS transpose fence (wave-private)

      // attn tile: 4x b128 non-temporal stores per lane (row-contiguous from LDS)
#pragma unroll
      for (int i = 0; i < 4; ++i) {
        const int idx = lane + 32 * i;
        const int row = idx >> 3;
        const int cg  = (idx & 7) << 2;
        const f32x4 t4 = *(const f32x4*)(lp + row * 32 + cg);
        __builtin_nontemporal_store(t4, (f32x4*)(attn + (bhS + qb + row) * S_LEN + kt32 + cg));
      }

      // P as A-fragment (bf16) from the same LDS tile
      v16bf pa;
#pragma unroll
      for (int vv = 0; vv < 8; ++vv) {
        const float2 t = *(const float2*)(lp + n * 32 + 16 * h4 + 2 * vv);
        bf16x2 c = pk_bf16(t.x, t.y);
        pa[2 * vv]     = c[0];
        pa[2 * vv + 1] = c[1];
      }
      const float* vb = vbase_bh + (size_t)(kt32 + 16 * h4) * D_DIM + n;
#pragma unroll
      for (int db = 0; db < 4; ++db) {
        o[db] = __builtin_amdgcn_wmma_f32_16x16x32_bf16(false, pa, false, load_v_frag(vb + db * 16),
                                                        (short)0, o[db], false, false);
      }
    } else {
      // fully-masked region: softmax of -1e9 underflows to exact 0.0f -> b128 zero fill
      const f32x4 z4 = {0.0f, 0.0f, 0.0f, 0.0f};
#pragma unroll
      for (int i = 0; i < 4; ++i) {
        const int idx = lane + 32 * i;
        const int row = idx >> 3;
        const int cg  = (idx & 7) << 2;
        __builtin_nontemporal_store(z4, (f32x4*)(attn + (bhS + qb + row) * S_LEN + kt32 + cg));
      }
    }
  }

  // ---- write O (16x64 tile) ----
#pragma unroll
  for (int db = 0; db < 4; ++db)
#pragma unroll
    for (int r = 0; r < 8; ++r)
      __builtin_nontemporal_store(o[db][r],
          outp + (bhS + qb + r + 8 * h4) * D_DIM + db * 16 + n);
}

extern "C" void kernel_launch(void* const* d_in, const int* in_sizes, int n_in,
                              void* d_out, int out_size, void* d_ws, size_t ws_size,
                              hipStream_t stream) {
  const float* q = (const float*)d_in[0];
  const float* k = (const float*)d_in[1];
  const float* v = (const float*)d_in[2];
  // d_in[3] (mask) ignored: causal mask applied analytically.
  float* outp = (float*)d_out;
  float* attn = outp + (size_t)B_N * H_N * S_LEN * D_DIM;

  const int total_tiles = B_N * H_N * (S_LEN / 16);  // 4096
  dim3 grid(total_tiles / 8);                        // 8 waves (q-tiles) per block
  dim3 block(256);
  hipLaunchKernelGGL(fa_causal_kernel, grid, block, 0, stream, q, k, v, outp, attn);
}
